// AttentionBlock_68341519614205
// MI455X (gfx1250) — compile-verified, matrix operand order checked by reading, not run on hardware
//
#include <hip/hip_runtime.h>

typedef __attribute__((ext_vector_type(16))) _Float16 v16h;
typedef __attribute__((ext_vector_type(8)))  _Float16 v8h;
typedef __attribute__((ext_vector_type(8)))  float    v8f;
typedef __attribute__((address_space(3)))    void     lds_void;

// ---------------------------------------------------------------------------
// Problem constants (B=16, C=256, H=W=32, N=1024, GROUPS=8)
// ---------------------------------------------------------------------------
#define BATCH 16
#define CH    256
#define NPIX  1024
#define NQKV  768

__device__ __forceinline__ v8f wmma_f16(v16h a, v16h b, v8f c) {
  // D = A(16x32 f16) * B(32x16 f16) + C(16x16 f32)
  return __builtin_amdgcn_wmma_f32_16x16x32_f16(
      /*neg_a=*/false, a, /*neg_b=*/false, b,
      /*c_mod=*/(short)0, c, /*reuse_a=*/false, /*reuse_b=*/false);
}

// Fragment loader matching CDNA5 16-bit A/B WMMA layouts:
// lane = r + 16*h ; VGPR0..3 hold K = h*8+{0..7}, VGPR4..7 hold K = 16+h*8+{0..7}
// For A: r = M row.  For B: r = N column, rows of `tile` indexed by N, contiguous K.
__device__ __forceinline__ v16h frag_ld(const _Float16* tile, int pitch, int lane) {
  const int r = lane & 15;
  const int h = (lane >> 4) & 1;
  const _Float16* p = tile + r * pitch + h * 8;
  v16h f;
  ((v8h*)&f)[0] = *(const v8h*)(p);
  ((v8h*)&f)[1] = *(const v8h*)(p + 16);
  return f;
}

__device__ __forceinline__ void cp16(_Float16* dst, const _Float16* src) {
  *(v8h*)dst = *(const v8h*)src;   // one b128 transaction
}

// Async DMA: 16 bytes global -> LDS, tracked by ASYNCcnt (no VGPR round-trip).
__device__ __forceinline__ void async_cp16(void* lds_dst, const void* gsrc) {
  const unsigned loff = (unsigned)(unsigned long long)(lds_void*)lds_dst;
  const unsigned long long ga = (unsigned long long)gsrc;
  asm volatile("global_load_async_to_lds_b128 %0, %1, off"
               :
               : "v"(loff), "v"(ga)
               : "memory");
}
__device__ __forceinline__ void async_wait0() {
  asm volatile("s_wait_asynccnt 0x0" ::: "memory");
}

// ---------------------------------------------------------------------------
// Kernel 1: GroupNorm statistics. One block per (batch, group).
// ---------------------------------------------------------------------------
__global__ __launch_bounds__(256) void gn_stats_kernel(const float* __restrict__ x,
                                                       float* __restrict__ stat) {
  const int b = blockIdx.x >> 3;
  const int g = blockIdx.x & 7;
  const float* base = x + ((size_t)b * CH + (size_t)g * 32) * NPIX;
  float s = 0.f, s2 = 0.f;
  for (int i = threadIdx.x; i < 32 * NPIX; i += 256) {
    float v = base[i];
    s  += v;
    s2 += v * v;
  }
  __shared__ float rs[256];
  __shared__ float rq[256];
  rs[threadIdx.x] = s;
  rq[threadIdx.x] = s2;
  __syncthreads();
  for (int off = 128; off > 0; off >>= 1) {
    if (threadIdx.x < off) {
      rs[threadIdx.x] += rs[threadIdx.x + off];
      rq[threadIdx.x] += rq[threadIdx.x + off];
    }
    __syncthreads();
  }
  if (threadIdx.x == 0) {
    const float inv_n = 1.0f / (32.0f * (float)NPIX);
    float mean = rs[0] * inv_n;
    float var  = rq[0] * inv_n - mean * mean;
    stat[2 * blockIdx.x + 0] = mean;
    stat[2 * blockIdx.x + 1] = rsqrtf(var + 1e-5f);
  }
}

// ---------------------------------------------------------------------------
// Kernel 2: normalize + gamma/beta, write f16 as xn[b][pixel][c]
// ---------------------------------------------------------------------------
__global__ __launch_bounds__(256) void gn_apply_kernel(const float* __restrict__ x,
                                                       const float* __restrict__ gamma,
                                                       const float* __restrict__ beta,
                                                       const float* __restrict__ stat,
                                                       _Float16* __restrict__ xn) {
  const int idx = blockIdx.x * 256 + threadIdx.x;  // over B*C*N
  const int b = idx >> 18;          // / (CH*NPIX)
  const int c = (idx >> 10) & 255;
  const int p = idx & 1023;
  const int sg = (b << 3) | (c >> 5);
  const float mean = stat[2 * sg + 0];
  const float rstd = stat[2 * sg + 1];
  const float v = (x[idx] - mean) * rstd * gamma[c] + beta[c];
  xn[((size_t)b * NPIX + p) * CH + c] = (_Float16)v;
}

// ---------------------------------------------------------------------------
// Kernel 3: weights fp32 -> f16 (layout kept [o][k] == WMMA B n-major layout)
// ---------------------------------------------------------------------------
__global__ __launch_bounds__(256) void cvt_w_kernel(const float* __restrict__ wq,
                                                    const float* __restrict__ wp,
                                                    _Float16* __restrict__ wqh,
                                                    _Float16* __restrict__ wph) {
  const int i = blockIdx.x * 256 + threadIdx.x;
  if (i < NQKV * CH) wqh[i] = (_Float16)wq[i];
  if (i < CH * CH)   wph[i] = (_Float16)wp[i];
}

// ---------------------------------------------------------------------------
// Kernel 4: QKV GEMM.  Y[n][o] = sum_k xn[n][k] * w[o][k] + bias[o]
// Block tile 128(M) x 64(N), 8 waves in a 4x2 grid, each wave 32x32 (4 accums)
// Q scaled by 1/sqrt(C); V stored transposed vt[b][c][pixel].
// ---------------------------------------------------------------------------
#define PQ 40   // LDS pitch (f16): 80B rows, 16B aligned, conflict-free
__global__ __launch_bounds__(256) void gemm_qkv_kernel(const _Float16* __restrict__ xn,
                                                       const _Float16* __restrict__ wq,
                                                       const float* __restrict__ bqkv,
                                                       _Float16* __restrict__ qf,
                                                       _Float16* __restrict__ kf,
                                                       _Float16* __restrict__ vt) {
  __shared__ _Float16 As[128 * PQ];
  __shared__ _Float16 Bs[64 * PQ];
  const int b  = blockIdx.z;
  const int mB = blockIdx.y * 128;
  const int nB = blockIdx.x * 64;
  const int tid = threadIdx.x, lane = tid & 31, wave = tid >> 5;
  const int wm = wave >> 1, wn = wave & 1;
  const int h = lane >> 4, cl = lane & 15;
  const _Float16* Ag = xn + ((size_t)b * NPIX + mB) * CH;

  v8f acc[2][2];
#pragma unroll
  for (int mi = 0; mi < 2; ++mi)
#pragma unroll
    for (int ni = 0; ni < 2; ++ni)
#pragma unroll
      for (int e = 0; e < 8; ++e) acc[mi][ni][e] = 0.f;

  for (int k0 = 0; k0 < CH; k0 += 32) {
    {  // stage A tile 128x32 via async DMA to LDS
      const int r = tid >> 1, s = tid & 1;
      const _Float16* src = Ag + (size_t)r * CH + k0 + s * 16;
      async_cp16(&As[r * PQ + s * 16],     src);
      async_cp16(&As[r * PQ + s * 16 + 8], src + 8);
      if (k0 + 32 < CH) __builtin_prefetch(src + 32, 0, 3);
    }
    {  // stage B tile 64x32 (rows = output channel, contiguous k)
      const int r = tid >> 2, s = tid & 3;
      async_cp16(&Bs[r * PQ + s * 8], wq + (size_t)(nB + r) * CH + k0 + s * 8);
    }
    async_wait0();
    __syncthreads();
    v16h a0 = frag_ld(&As[(wm * 32 + 0)  * PQ], PQ, lane);
    v16h a1 = frag_ld(&As[(wm * 32 + 16) * PQ], PQ, lane);
    v16h b0 = frag_ld(&Bs[(wn * 32 + 0)  * PQ], PQ, lane);
    v16h b1 = frag_ld(&Bs[(wn * 32 + 16) * PQ], PQ, lane);
    acc[0][0] = wmma_f16(a0, b0, acc[0][0]);
    acc[0][1] = wmma_f16(a0, b1, acc[0][1]);
    acc[1][0] = wmma_f16(a1, b0, acc[1][0]);
    acc[1][1] = wmma_f16(a1, b1, acc[1][1]);
    __syncthreads();
  }

  // Epilogue: D lane layout -> lane holds rows m = 8*h + r, column cl.
#pragma unroll
  for (int mi = 0; mi < 2; ++mi)
#pragma unroll
    for (int ni = 0; ni < 2; ++ni) {
      const int o    = nB + wn * 32 + ni * 16 + cl;       // segment-uniform per tile
      const int prow = mB + wm * 32 + mi * 16 + h * 8;
      const float bias = bqkv[o];
      if (o < 512) {
        _Float16* dst = (o < 256) ? qf : kf;
        const int   oc    = o & 255;
        const float scale = (o < 256) ? 0.0625f : 1.0f;   // q *= 1/sqrt(256)
#pragma unroll
        for (int r = 0; r < 8; ++r) {
          float v = (acc[mi][ni][r] + bias) * scale;
          dst[((size_t)b * NPIX + prow + r) * CH + oc] = (_Float16)v;
        }
      } else {
        const int oc = o - 512;                            // V: store transposed
        v8h pk;
#pragma unroll
        for (int r = 0; r < 8; ++r) pk[r] = (_Float16)(acc[mi][ni][r] + bias);
        *(v8h*)&vt[((size_t)b * CH + oc) * NPIX + prow] = pk;  // 8 contiguous pixels
      }
    }
}

// ---------------------------------------------------------------------------
// Kernel 5: flash attention.  One block = (batch, 128 query rows); 8 waves,
// each wave owns 16 query rows.  Key blocks of 32 staged in LDS via async DMA.
// ---------------------------------------------------------------------------
#define PK 264  // K-tile pitch (256 + 8)
#define PV 40   // V/P-tile pitch (32 + 8)
__global__ __launch_bounds__(256) void attn_kernel(const _Float16* __restrict__ qf,
                                                   const _Float16* __restrict__ kf,
                                                   const _Float16* __restrict__ vt,
                                                   _Float16* __restrict__ of) {
  __shared__ _Float16 Ks[32 * PK];        // [key][c]
  __shared__ _Float16 Vs[256 * PV];       // [c][key]
  __shared__ _Float16 Ps[8][16 * PV];     // per-wave P scratch [row][key]
  const int b  = blockIdx.x >> 3;
  const int mB = (blockIdx.x & 7) * 128;
  const int tid = threadIdx.x, lane = tid & 31, wave = tid >> 5;
  const int h = lane >> 4, cl = lane & 15;

  const _Float16* Aq = qf + ((size_t)b * NPIX + mB + wave * 16) * CH;

  float mrow[8], lrow[8];
  v8f accO[16];
#pragma unroll
  for (int r = 0; r < 8; ++r) { mrow[r] = -1e30f; lrow[r] = 0.f; }
#pragma unroll
  for (int t = 0; t < 16; ++t)
#pragma unroll
    for (int e = 0; e < 8; ++e) accO[t][e] = 0.f;

  for (int kb = 0; kb < 32; ++kb) {       // 32 key blocks of 32 keys
    {  // stage K tile 32x256 via async DMA
      const int r = tid >> 3, s = tid & 7;
      const _Float16* src = kf + ((size_t)b * NPIX + kb * 32 + r) * CH + s * 32;
      _Float16* dst = &Ks[r * PK + s * 32];
#pragma unroll
      for (int i = 0; i < 32; i += 8) async_cp16(dst + i, src + i);
    }
    {  // stage V tile 256x32 (already [c][pixel] in global) via async DMA
      const _Float16* src = vt + ((size_t)b * CH + tid) * NPIX + kb * 32;
      _Float16* dst = &Vs[tid * PV];
#pragma unroll
      for (int i = 0; i < 32; i += 8) async_cp16(dst + i, src + i);
    }
    async_wait0();
    __syncthreads();

    // S = Q(16xC) * K^T(Cx32)  -> 2 accum tiles of 16x16
    v8f s[2];
#pragma unroll
    for (int nt = 0; nt < 2; ++nt)
#pragma unroll
      for (int e = 0; e < 8; ++e) s[nt][e] = 0.f;
    for (int k0 = 0; k0 < CH; k0 += 32) {
      v16h af = frag_ld(Aq + k0, CH, lane);            // Q fragment from L2
      v16h bf0 = frag_ld(&Ks[0 * PK + k0],  PK, lane);
      v16h bf1 = frag_ld(&Ks[16 * PK + k0], PK, lane);
      s[0] = wmma_f16(af, bf0, s[0]);
      s[1] = wmma_f16(af, bf1, s[1]);
    }

    // Flash softmax update. Lane holds rows m=8*h+r, col cl (+16*nt).
#pragma unroll
    for (int r = 0; r < 8; ++r) {
      float mx = fmaxf(s[0][r], s[1][r]);
#pragma unroll
      for (int off = 1; off < 16; off <<= 1) mx = fmaxf(mx, __shfl_xor(mx, off, 32));
      const float mnew = fmaxf(mrow[r], mx);
      const float corr = __expf(mrow[r] - mnew);
      float ls = 0.f;
#pragma unroll
      for (int nt = 0; nt < 2; ++nt) {
        float p = __expf(s[nt][r] - mnew);
        s[nt][r] = p;
        ls += p;
      }
#pragma unroll
      for (int off = 1; off < 16; off <<= 1) ls += __shfl_xor(ls, off, 32);
      lrow[r] = lrow[r] * corr + ls;
      mrow[r] = mnew;
#pragma unroll
      for (int t = 0; t < 16; ++t) accO[t][r] *= corr;
    }

    // Reshape P: D layout -> A layout via per-wave LDS scratch
#pragma unroll
    for (int nt = 0; nt < 2; ++nt)
#pragma unroll
      for (int r = 0; r < 8; ++r)
        Ps[wave][(h * 8 + r) * PV + nt * 16 + cl] = (_Float16)s[nt][r];
    __syncthreads();

    // O += P(16x32) * V(32x256): 16 channel tiles, one K step
    {
      v16h af = frag_ld(&Ps[wave][0], PV, lane);
#pragma unroll
      for (int ct = 0; ct < 16; ++ct) {
        v16h bf = frag_ld(&Vs[(ct * 16) * PV], PV, lane);
        accO[ct] = wmma_f16(af, bf, accO[ct]);
      }
    }
    __syncthreads();
  }

  // Normalize and store O as of[b][pixel][c] (f16)
#pragma unroll
  for (int r = 0; r < 8; ++r) {
    const float inv = 1.0f / lrow[r];
    const int pix = mB + wave * 16 + h * 8 + r;
    _Float16* dst = of + ((size_t)b * NPIX + pix) * CH;
#pragma unroll
    for (int ct = 0; ct < 16; ++ct)
      dst[ct * 16 + cl] = (_Float16)(accO[ct][r] * inv);
  }
}

// ---------------------------------------------------------------------------
// Kernel 6: proj GEMM + bias + residual.  out[b][c][pixel] (fp32)
// ---------------------------------------------------------------------------
__global__ __launch_bounds__(256) void gemm_proj_kernel(const _Float16* __restrict__ of,
                                                        const _Float16* __restrict__ wp,
                                                        const float* __restrict__ bproj,
                                                        const float* __restrict__ xres,
                                                        float* __restrict__ out) {
  __shared__ _Float16 As[128 * PQ];
  __shared__ _Float16 Bs[64 * PQ];
  const int b  = blockIdx.z;
  const int mB = blockIdx.y * 128;
  const int nB = blockIdx.x * 64;
  const int tid = threadIdx.x, lane = tid & 31, wave = tid >> 5;
  const int wm = wave >> 1, wn = wave & 1;
  const int h = lane >> 4, cl = lane & 15;
  const _Float16* Ag = of + ((size_t)b * NPIX + mB) * CH;

  v8f acc[2][2];
#pragma unroll
  for (int mi = 0; mi < 2; ++mi)
#pragma unroll
    for (int ni = 0; ni < 2; ++ni)
#pragma unroll
      for (int e = 0; e < 8; ++e) acc[mi][ni][e] = 0.f;

  for (int k0 = 0; k0 < CH; k0 += 32) {
    {
      const int r = tid >> 1, s = tid & 1;
      const _Float16* src = Ag + (size_t)r * CH + k0 + s * 16;
      async_cp16(&As[r * PQ + s * 16],     src);
      async_cp16(&As[r * PQ + s * 16 + 8], src + 8);
    }
    {
      const int r = tid >> 2, s = tid & 3;
      async_cp16(&Bs[r * PQ + s * 8], wp + (size_t)(nB + r) * CH + k0 + s * 8);
    }
    async_wait0();
    __syncthreads();
    v16h a0 = frag_ld(&As[(wm * 32 + 0)  * PQ], PQ, lane);
    v16h a1 = frag_ld(&As[(wm * 32 + 16) * PQ], PQ, lane);
    v16h b0 = frag_ld(&Bs[(wn * 32 + 0)  * PQ], PQ, lane);
    v16h b1 = frag_ld(&Bs[(wn * 32 + 16) * PQ], PQ, lane);
    acc[0][0] = wmma_f16(a0, b0, acc[0][0]);
    acc[0][1] = wmma_f16(a0, b1, acc[0][1]);
    acc[1][0] = wmma_f16(a1, b0, acc[1][0]);
    acc[1][1] = wmma_f16(a1, b1, acc[1][1]);
    __syncthreads();
  }

#pragma unroll
  for (int mi = 0; mi < 2; ++mi)
#pragma unroll
    for (int ni = 0; ni < 2; ++ni) {
      const int chan = nB + wn * 32 + ni * 16 + cl;
      const int pix0 = mB + wm * 32 + mi * 16 + h * 8;
      const float bias = bproj[chan];
      const size_t base = ((size_t)b * CH + chan) * NPIX + pix0;
#pragma unroll
      for (int r = 0; r < 8; ++r)
        out[base + r] = acc[mi][ni][r] + bias + xres[base + r];
    }
}

// ---------------------------------------------------------------------------
// Host launcher
// ---------------------------------------------------------------------------
extern "C" void kernel_launch(void* const* d_in, const int* in_sizes, int n_in,
                              void* d_out, int out_size, void* d_ws, size_t ws_size,
                              hipStream_t stream) {
  (void)in_sizes; (void)n_in; (void)out_size; (void)ws_size;
  const float* x      = (const float*)d_in[0];
  const float* gamma  = (const float*)d_in[1];
  const float* beta   = (const float*)d_in[2];
  const float* w_qkv  = (const float*)d_in[3];
  const float* b_qkv  = (const float*)d_in[4];
  const float* w_proj = (const float*)d_in[5];
  const float* b_proj = (const float*)d_in[6];
  float* out = (float*)d_out;

  char* ws = (char*)d_ws;
  auto carve = [&](size_t bytes) -> char* {
    char* p = ws;
    ws += (bytes + 255) & ~(size_t)255;
    return p;
  };
  const size_t act = (size_t)BATCH * NPIX * CH * sizeof(_Float16);  // 8 MB
  float*    stat = (float*)   carve(BATCH * 8 * 2 * sizeof(float));
  _Float16* xn   = (_Float16*)carve(act);
  _Float16* wqh  = (_Float16*)carve((size_t)NQKV * CH * sizeof(_Float16));
  _Float16* wph  = (_Float16*)carve((size_t)CH * CH * sizeof(_Float16));
  _Float16* qf   = (_Float16*)carve(act);
  _Float16* kf   = (_Float16*)carve(act);
  _Float16* vt   = (_Float16*)carve(act);
  _Float16* of   = (_Float16*)carve(act);

  gn_stats_kernel<<<BATCH * 8, 256, 0, stream>>>(x, stat);
  gn_apply_kernel<<<(BATCH * CH * NPIX) / 256, 256, 0, stream>>>(x, gamma, beta, stat, xn);
  cvt_w_kernel<<<(NQKV * CH) / 256, 256, 0, stream>>>(w_qkv, w_proj, wqh, wph);
  gemm_qkv_kernel<<<dim3(NQKV / 64, NPIX / 128, BATCH), 256, 0, stream>>>(
      xn, wqh, b_qkv, qf, kf, vt);
  attn_kernel<<<BATCH * (NPIX / 128), 256, 0, stream>>>(qf, kf, vt, of);
  gemm_proj_kernel<<<dim3(CH / 64, NPIX / 128, BATCH), 256, 0, stream>>>(
      of, wph, b_proj, x, out);
}
